// SparseCnnDown1_52561809768823
// MI455X (gfx1250) — compile-verified, matrix operand order checked by reading, not run on hardware
//
#include <hip/hip_runtime.h>

// CDNA5 / gfx1250: wave32, WMMA bf16 16x16x32, f32 accumulate.
typedef __attribute__((ext_vector_type(16))) __bf16 v16bf;
typedef __attribute__((ext_vector_type(8)))  float  v8f;

// f32 -> bf16 (RNE via hardware cvt), packed pair into a dword.
__device__ inline unsigned short f2bf(float x) {
  union { __bf16 h; unsigned short u; } cv;
  cv.h = (__bf16)x;
  return cv.u;
}
__device__ inline unsigned pack2(float lo, float hi) {
  return (unsigned)f2bf(lo) | ((unsigned)f2bf(hi) << 16);
}

// Fragment-order addressing (16-bit A/B, 16x32 chunk):
//   element (m|n, k):  lane = (m & 15) + 16*((k>>3)&1)
//                      idx  = (k & 7) + 8*(k>>4)          (16 bf16 per lane)
// One 32-lane chunk = 512 bf16 = 256 dwords.

// ---------------------------------------------------------------------------
// Elementwise helpers
// ---------------------------------------------------------------------------
__global__ void fill_bias_kernel(float* __restrict__ out, const float* __restrict__ b,
                                 int n, int C) {
  int i = blockIdx.x * blockDim.x + threadIdx.x;
  if (i < n * C) out[i] = b[i % C];
}

__global__ void relu_kernel(float* __restrict__ x, int n) {
  int i = blockIdx.x * blockDim.x + threadIdx.x;
  if (i < n) x[i] = fmaxf(x[i], 0.f);
}

// x[n2,64] += cat(a[n2,32], b[n2,32])
__global__ void combine_kernel(float* __restrict__ x, const float* __restrict__ a,
                               const float* __restrict__ b, int n2) {
  int i = blockIdx.x * blockDim.x + threadIdx.x;
  if (i >= n2 * 64) return;
  int row = i >> 6, c = i & 63;
  float add = (c < 32) ? a[row * 32 + c] : b[row * 32 + (c - 32)];
  x[i] += add;
}

// ---------------------------------------------------------------------------
// Weight pre-pack: f32 [K, CIN, COUT] -> bf16 B-fragments
// dst dword u = ((tap*NCT + ct)*NCH + ch)*256 + lane*8 + idxpair
// ---------------------------------------------------------------------------
template <int CIN, int COUT>
__global__ void pack_w_kernel(const float* __restrict__ w, int K,
                              unsigned* __restrict__ dst) {
  constexpr int NCH = (CIN + 31) / 32;
  constexpr int NCT = COUT / 16;
  const int total = K * NCT * NCH * 256;
  int u = blockIdx.x * blockDim.x + threadIdx.x;
  if (u >= total) return;
  int idxp = u & 7;
  int lane = (u >> 3) & 31;
  int ch   = (u >> 8) % NCH;
  int ct   = ((u >> 8) / NCH) % NCT;
  int tap  = u / (256 * NCH * NCT);
  int n = lane & 15, hi = lane >> 4;
  int idx0 = idxp * 2;
  int k0  = (idx0 & 7) + 8 * hi + 16 * (idx0 >> 3);
  int cin = ch * 32 + k0;
  int cout = ct * 16 + n;
  float v0 = (cin     < CIN) ? w[((long long)tap * CIN + cin)     * COUT + cout] : 0.f;
  float v1 = (cin + 1 < CIN) ? w[((long long)tap * CIN + cin + 1) * COUT + cout] : 0.f;
  dst[u] = pack2(v0, v1);
}

// ---------------------------------------------------------------------------
// Sparse conv tap-GEMM.  Block = 128 threads = 4 wave32 per (tap, 64-pair
// tile = 4 M-subtiles of 16).  kin/kout indices staged to LDS once; 64
// gathered rows staged as bf16 A-fragments in LDS.  Waves split
// (ct = waveId%NCT, subtile) so all 4 wave32 are busy for any COUT; B
// fragments live in registers and are reused across a wave's subtiles.
// Scatter via f32 hardware atomics (L2-resident).
// ---------------------------------------------------------------------------
template <int CIN, int COUT>
__launch_bounds__(128)
__global__ void sparse_conv_wmma_kernel(
    const float* __restrict__ feats, int n_in,
    const unsigned* __restrict__ wp,
    const int* __restrict__ kin, const int* __restrict__ kout,
    int n_out, int L, float* __restrict__ out)
{
  constexpr int NCH = (CIN + 31) / 32;
  constexpr int NCT = COUT / 16;
  constexpr int NST = 4;                       // M-subtiles per block
  __shared__ int      sRow[64];
  __shared__ int      sOut[64];
  __shared__ unsigned sA[NST * NCH * 256];

  const int tap = blockIdx.y, tile = blockIdx.x;
  const int tid = threadIdx.x, lane = tid & 31, waveId = tid >> 5;
  const int tileBase = tile * 64;
  const long long tapL = (long long)tap * L;

  if (tid < 64) {
    int p = tileBase + tid;
    sRow[tid] = (p < L) ? kin[tapL + p]  : n_in;   // sentinel -> zero row
    sOut[tid] = (p < L) ? kout[tapL + p] : -1;     // sentinel -> no scatter
  }
  __syncthreads();

  // Stage A fragments for all 4 subtiles (bf16 pairs, ds_store_b32).
  for (int u = tid; u < NST * NCH * 256; u += 128) {
    int e    = u & 255;
    int ch   = (u >> 8) % NCH;
    int st   = (u >> 8) / NCH;
    int idxp = e & 7;
    int l2   = (e >> 3) & 31;
    int m = l2 & 15, hi = l2 >> 4;
    int idx0 = idxp * 2;
    int k0  = (idx0 & 7) + 8 * hi + 16 * (idx0 >> 3);
    int col = ch * 32 + k0;
    int r = sRow[st * 16 + m];
    float v0 = 0.f, v1 = 0.f;
    if (r >= 0 && r < n_in) {
      const float* fr = feats + (long long)r * CIN;
      if (col     < CIN) v0 = fr[col];
      if (col + 1 < CIN) v1 = fr[col + 1];
    }
    sA[u] = pack2(v0, v1);
  }
  __syncthreads();

  // B fragments: loaded once per wave, reused across subtiles.
  const int ct = waveId % NCT;
  const unsigned* wt = wp + ((long long)(tap * NCT + ct) * NCH) * 256;
  v16bf bfrag[NCH];
#pragma unroll
  for (int ch = 0; ch < NCH; ++ch)
    bfrag[ch] = *(const v16bf*)&wt[ch * 256 + lane * 8];

  const int mn = lane & 15, mBase = (lane >> 4) << 3;
  const int st0 = waveId / NCT;
#pragma unroll
  for (int i = 0; i < NCT; ++i) {             // constant trip count -> unrolls
    const int st = st0 + i * (4 / NCT);
    v8f acc = {};
#pragma unroll
    for (int ch = 0; ch < NCH; ++ch) {
      v16bf a = *(const v16bf*)&sA[(st * NCH + ch) * 256 + lane * 8];
      acc = __builtin_amdgcn_wmma_f32_16x16x32_bf16(false, a, false, bfrag[ch],
                                                    (short)0, acc, false, false);
    }
    // D fragment: VGPR r holds M = r + (lane<16 ? 0 : 8), N = lane&15.
#pragma unroll
    for (int r = 0; r < 8; ++r) {
      int o = sOut[st * 16 + mBase + r];
      if (o >= 0 && o < n_out)
        unsafeAtomicAdd(&out[(long long)o * COUT + ct * 16 + mn], acc[r]);
    }
  }
}

// ---------------------------------------------------------------------------
// Dense GEMM (1x1 convs): identity row map, direct store, fused bias+ReLU.
// Same 64-row tiling and wave work split as the sparse kernel.
// ---------------------------------------------------------------------------
template <int CIN, int COUT>
__launch_bounds__(128)
__global__ void dense_gemm_wmma_kernel(
    const float* __restrict__ x, int nRows,
    const unsigned* __restrict__ wp, const float* __restrict__ bias,
    float* __restrict__ out, int doRelu)
{
  constexpr int NCH = (CIN + 31) / 32;
  constexpr int NCT = COUT / 16;
  constexpr int NST = 4;
  __shared__ unsigned sA[NST * NCH * 256];
  const int tile = blockIdx.x;
  const int tid = threadIdx.x, lane = tid & 31, waveId = tid >> 5;
  const int tileBase = tile * 64;

  for (int u = tid; u < NST * NCH * 256; u += 128) {
    int e    = u & 255;
    int ch   = (u >> 8) % NCH;
    int st   = (u >> 8) / NCH;
    int idxp = e & 7;
    int l2   = (e >> 3) & 31;
    int m = l2 & 15, hi = l2 >> 4;
    int idx0 = idxp * 2;
    int k0  = (idx0 & 7) + 8 * hi + 16 * (idx0 >> 3);
    int col = ch * 32 + k0;
    int p = tileBase + st * 16 + m;
    float v0 = 0.f, v1 = 0.f;
    if (p < nRows) {
      const float* xr = x + (long long)p * CIN;
      if (col     < CIN) v0 = xr[col];
      if (col + 1 < CIN) v1 = xr[col + 1];
    }
    sA[u] = pack2(v0, v1);
  }
  __syncthreads();

  const int ct = waveId % NCT;
  const unsigned* wt = wp + (long long)ct * NCH * 256;
  v16bf bfrag[NCH];
#pragma unroll
  for (int ch = 0; ch < NCH; ++ch)
    bfrag[ch] = *(const v16bf*)&wt[ch * 256 + lane * 8];

  const int mn = lane & 15, mBase = (lane >> 4) << 3;
  const float bv = bias[ct * 16 + mn];
  const int st0 = waveId / NCT;
#pragma unroll
  for (int i = 0; i < NCT; ++i) {
    const int st = st0 + i * (4 / NCT);
    v8f acc = {};
#pragma unroll
    for (int ch = 0; ch < NCH; ++ch) {
      v16bf a = *(const v16bf*)&sA[(st * NCH + ch) * 256 + lane * 8];
      acc = __builtin_amdgcn_wmma_f32_16x16x32_bf16(false, a, false, bfrag[ch],
                                                    (short)0, acc, false, false);
    }
#pragma unroll
    for (int r = 0; r < 8; ++r) {
      int p = tileBase + st * 16 + mBase + r;
      if (p < nRows) {
        float v = acc[r] + bv;
        if (doRelu) v = fmaxf(v, 0.f);
        out[(long long)p * COUT + ct * 16 + mn] = v;
      }
    }
  }
}

// ---------------------------------------------------------------------------
// Host-side pipeline
// ---------------------------------------------------------------------------
static inline size_t packed_uints(int K, int CIN, int COUT) {
  return (size_t)K * (COUT / 16) * ((CIN + 31) / 32) * 256;
}

extern "C" void kernel_launch(void* const* d_in, const int* in_sizes, int n_in,
                              void* d_out, int out_size, void* d_ws, size_t ws_size,
                              hipStream_t stream)
{
  (void)out_size; (void)ws_size;

  // Index maps for the two plausible flattenings of setup_inputs().
  // Layout A (dict insertion order):  in_sizes[1] == 27*32*32 == 27648  (w_a)
  // Layout B (recursive sorted keys): in_sizes[1] == 27*131072          (kin1)
  int I_feats, I_wa, I_ba, I_wd, I_bd, I_wlast, I_blast;
  int I_irn[3][10];  // order: w00,b00,w01,b01,w10,b10,w11,b11,w12,b12
  int I_kin1, I_kout1, I_kind, I_koutd, I_kin2, I_kout2;

  const bool insertionOrder = (n_in > 1 && in_sizes[1] == 27648);
  if (insertionOrder) {
    I_feats = 0;
    I_wa = 1; I_ba = 2; I_wd = 3; I_bd = 4; I_wlast = 5; I_blast = 6;
    for (int i = 0; i < 3; ++i)
      for (int j = 0; j < 10; ++j) I_irn[i][j] = 7 + i * 10 + j;
    I_kin1 = 37; I_kout1 = 38; I_kind = 39; I_koutd = 40; I_kin2 = 41; I_kout2 = 42;
  } else {
    I_feats = 0; I_kin1 = 1; I_kin2 = 2; I_kind = 3;
    I_kout1 = 4; I_kout2 = 5; I_koutd = 6;
    I_ba = 8; I_bd = 9; I_blast = 10;
    for (int i = 0; i < 3; ++i) {
      int base = 11 + i * 10;   // b00,b01,b10,b11,b12,w00,w01,w10,w11,w12
      I_irn[i][0] = base + 5; I_irn[i][1] = base + 0;
      I_irn[i][2] = base + 6; I_irn[i][3] = base + 1;
      I_irn[i][4] = base + 7; I_irn[i][5] = base + 2;
      I_irn[i][6] = base + 8; I_irn[i][7] = base + 3;
      I_irn[i][8] = base + 9; I_irn[i][9] = base + 4;
    }
    I_wa = 41; I_wd = 42; I_wlast = 43;
  }

  const int N1 = in_sizes[I_feats] / 32;   // feats: [N1, 32]
  const int n2 = in_sizes[I_kin2] / 27;    // kin2:  [27, n2]

  const float* feats  = (const float*)d_in[I_feats];
  const float* w_a    = (const float*)d_in[I_wa];
  const float* b_a    = (const float*)d_in[I_ba];
  const float* w_d    = (const float*)d_in[I_wd];
  const float* b_d    = (const float*)d_in[I_bd];
  const float* w_last = (const float*)d_in[I_wlast];
  const float* b_last = (const float*)d_in[I_blast];
  const int* kin1  = (const int*)d_in[I_kin1];
  const int* kout1 = (const int*)d_in[I_kout1];
  const int* kind  = (const int*)d_in[I_kind];
  const int* koutd = (const int*)d_in[I_koutd];
  const int* kin2  = (const int*)d_in[I_kin2];
  const int* kout2 = (const int*)d_in[I_kout2];

  // ---- workspace carve-up: activations (f32), then packed weights (u32) ----
  float* ws = (float*)d_ws;
  size_t off = 0;
  float* x1 = ws + off; off += (size_t)N1 * 32;
  float* x2 = ws + off; off += (size_t)n2 * 64;
  float* tA = ws + off; off += (size_t)n2 * 32;
  float* tB = ws + off; off += (size_t)n2 * 32;
  float* tC = ws + off; off += (size_t)n2 * 16;
  float* tD = ws + off; off += (size_t)n2 * 16;

  unsigned* up = (unsigned*)(ws + off);
  size_t uo = 0;
  unsigned* pw_a    = up + uo; uo += packed_uints(27, 32, 32);
  unsigned* pw_d    = up + uo; uo += packed_uints(8, 32, 64);
  unsigned* pw_last = up + uo; uo += packed_uints(27, 64, 64);
  unsigned* pw00[3]; unsigned* pw01[3]; unsigned* pw10[3];
  unsigned* pw11[3]; unsigned* pw12[3];
  for (int i = 0; i < 3; ++i) {
    pw00[i] = up + uo; uo += packed_uints(27, 64, 16);
    pw01[i] = up + uo; uo += packed_uints(27, 16, 32);
    pw10[i] = up + uo; uo += packed_uints(1, 64, 16);
    pw11[i] = up + uo; uo += packed_uints(27, 16, 16);
    pw12[i] = up + uo; uo += packed_uints(1, 16, 32);
  }

  // ---- pre-pack all weights to bf16 B-fragments ----
  auto packGrid = [](size_t n) { return (unsigned)((n + 255) / 256); };
  pack_w_kernel<32, 32><<<packGrid(packed_uints(27, 32, 32)), 256, 0, stream>>>(w_a, 27, pw_a);
  pack_w_kernel<32, 64><<<packGrid(packed_uints(8, 32, 64)), 256, 0, stream>>>(w_d, 8, pw_d);
  pack_w_kernel<64, 64><<<packGrid(packed_uints(27, 64, 64)), 256, 0, stream>>>(w_last, 27, pw_last);
  for (int i = 0; i < 3; ++i) {
    const float* w00 = (const float*)d_in[I_irn[i][0]];
    const float* w01 = (const float*)d_in[I_irn[i][2]];
    const float* w10 = (const float*)d_in[I_irn[i][4]];
    const float* w11 = (const float*)d_in[I_irn[i][6]];
    const float* w12 = (const float*)d_in[I_irn[i][8]];
    pack_w_kernel<64, 16><<<packGrid(packed_uints(27, 64, 16)), 256, 0, stream>>>(w00, 27, pw00[i]);
    pack_w_kernel<16, 32><<<packGrid(packed_uints(27, 16, 32)), 256, 0, stream>>>(w01, 27, pw01[i]);
    pack_w_kernel<64, 16><<<packGrid(packed_uints(1, 64, 16)), 256, 0, stream>>>(w10, 1, pw10[i]);
    pack_w_kernel<16, 16><<<packGrid(packed_uints(27, 16, 16)), 256, 0, stream>>>(w11, 27, pw11[i]);
    pack_w_kernel<16, 32><<<packGrid(packed_uints(1, 16, 32)), 256, 0, stream>>>(w12, 1, pw12[i]);
  }

  auto fill = [&](float* dst, const float* b, int n, int C) {
    int total = n * C;
    fill_bias_kernel<<<(total + 255) / 256, 256, 0, stream>>>(dst, b, n, C);
  };
  auto relu = [&](float* p, int n) {
    relu_kernel<<<(n + 255) / 256, 256, 0, stream>>>(p, n);
  };

  // --- down_block0 ---
  fill(x1, b_a, N1, 32);
  {
    dim3 g((N1 + 63) / 64, 27);
    sparse_conv_wmma_kernel<32, 32><<<g, 128, 0, stream>>>(
        feats, N1, pw_a, kin1, kout1, N1, N1, x1);
  }
  relu(x1, N1 * 32);

  fill(x2, b_d, n2, 64);
  {
    dim3 g((n2 + 63) / 64, 8);
    sparse_conv_wmma_kernel<32, 64><<<g, 128, 0, stream>>>(
        x1, N1, pw_d, kind, koutd, n2, n2, x2);
  }
  relu(x2, n2 * 64);

  // --- 3x InceptionResNet @ half res ---
  dim3 g2((n2 + 63) / 64, 27);
  int tiles2 = (n2 + 63) / 64;
  for (int i = 0; i < 3; ++i) {
    const float* b00 = (const float*)d_in[I_irn[i][1]];
    const float* b01 = (const float*)d_in[I_irn[i][3]];
    const float* b10 = (const float*)d_in[I_irn[i][5]];
    const float* b11 = (const float*)d_in[I_irn[i][7]];
    const float* b12 = (const float*)d_in[I_irn[i][9]];

    // branch0: tA = conv27( relu(conv27(x2, 64->16)), 16->32 )
    fill(tC, b00, n2, 16);
    sparse_conv_wmma_kernel<64, 16><<<g2, 128, 0, stream>>>(
        x2, n2, pw00[i], kin2, kout2, n2, n2, tC);
    relu(tC, n2 * 16);
    fill(tA, b01, n2, 32);
    sparse_conv_wmma_kernel<16, 32><<<g2, 128, 0, stream>>>(
        tC, n2, pw01[i], kin2, kout2, n2, n2, tA);

    // branch1: tB = (relu(conv27(relu(x2@w10+b10), 16->16)))@w12 + b12
    dense_gemm_wmma_kernel<64, 16><<<tiles2, 128, 0, stream>>>(
        x2, n2, pw10[i], b10, tD, /*relu=*/1);
    fill(tC, b11, n2, 16);
    sparse_conv_wmma_kernel<16, 16><<<g2, 128, 0, stream>>>(
        tD, n2, pw11[i], kin2, kout2, n2, n2, tC);
    relu(tC, n2 * 16);
    dense_gemm_wmma_kernel<16, 32><<<tiles2, 128, 0, stream>>>(
        tC, n2, pw12[i], b12, tB, /*relu=*/0);

    // x2 = cat(tA, tB) + x2
    combine_kernel<<<(n2 * 64 + 255) / 256, 256, 0, stream>>>(x2, tA, tB, n2);
  }

  // final ReLU then last conv27 C64->64 into d_out
  relu(x2, n2 * 64);
  float* out = (float*)d_out;
  fill(out, b_last, n2, 64);
  sparse_conv_wmma_kernel<64, 64><<<g2, 128, 0, stream>>>(
      x2, n2, pw_last, kin2, kout2, n2, n2, out);
}
